// NonOverlappingConv1d_shared_24885040513340
// MI455X (gfx1250) — matrix operand order repaired and checked by reading, not compiled, not run on hardware
//
#include <hip/hip_runtime.h>

typedef __attribute__((ext_vector_type(16))) _Float16 v16h;
typedef __attribute__((ext_vector_type(8)))  _Float16 v8h;
typedef __attribute__((ext_vector_type(8)))  float    v8f;
typedef __attribute__((ext_vector_type(4)))  float    v4f;

#define BS    32
#define CIN   128
#define DIN   4096
#define OC    256
#define POUT  1024          // DIN / 4
#define KDIM  256           // 2 * CIN
#define PAD   8
#define AK    (KDIM + PAD)  // 264-half row stride in LDS (conflict-free b128 loads)
#define MT    64            // o-chunk per A-stage
#define NT    128           // p-tile per workgroup
#define NMB   (OC / MT)     // 4

__global__ __launch_bounds__(256)
void conv_pairsum_wmma_kernel(const float* __restrict__ x,
                              const float* __restrict__ w,
                              float* __restrict__ out)
{
    extern __shared__ _Float16 smem[];
    _Float16* As  = smem;            // MT x AK halves (weight tile, f16)
    _Float16* Bsh = smem + MT * AK;  // NT x AK halves (pair-summed x tile, f16)

    const int tid  = threadIdx.x;
    const int lane = tid & 31;
    const int wv   = tid >> 5;

    const int b  = blockIdx.x >> 3;  // 32 batches
    const int nb = blockIdx.x & 7;   // 8 p-tiles
    const int p0 = nb * NT;

    // ---- Stage B once: Bsh[p][2c+k] = x[b,c,4(p0+p)+2k] + x[b,c,4(p0+p)+2k+1]
    {
        const v4f* x4   = (const v4f*)x;
        const int  rowq = DIN >> 2;                 // 1024 float4 per (b,c) row
        for (int i = tid; i < NT * CIN; i += 256) {
            const int c = i >> 7;                   // i / NT  (NT == 128)
            const int p = i & 127;                  // i % NT  -> coalesced float4
            v4f f = x4[(size_t)(b * CIN + c) * rowq + p0 + p];
            _Float16* dst = Bsh + p * AK + 2 * c;
            dst[0] = (_Float16)(f.x + f.y);         // k = 0
            dst[1] = (_Float16)(f.z + f.w);         // k = 1
        }
    }

    const float scale = 0.0883883476483184405f;     // 1/sqrt(128)

    for (int mb = 0; mb < NMB; ++mb) {
        const int o0 = mb * MT;

        __syncthreads();    // previous compute done before As overwrite
        // ---- Stage A: As[m][kk] = weight[(o0+m)*256 + kk]  (kk == 2c+k, contiguous!)
        {
            const v4f* w4 = (const v4f*)w;
            for (int i = tid; i < MT * (KDIM / 4); i += 256) {  // 4096 float4
                const int m = i >> 6;
                const int s = i & 63;
                v4f f = w4[(size_t)(o0 + m) * (KDIM / 4) + s];
                _Float16* dst = As + m * AK + 4 * s;
                dst[0] = (_Float16)f.x; dst[1] = (_Float16)f.y;
                dst[2] = (_Float16)f.z; dst[3] = (_Float16)f.w;
            }
        }
        __syncthreads();

        // ---- Compute: wave covers 32(o) x 32(p) = 2x2 WMMA tiles
        const int rm   = (wv & 1) * 32;   // 2 row groups  -> 64 o
        const int pn   = (wv >> 1) * 32;  // 4 col groups  -> 128 p
        const int nrow = lane & 15;       // N index / A-M row / B column
        const int hi16 = lane >> 4;       // lane group 0/1

        v8f acc[2][2] = {};

        #pragma unroll
        for (int kk = 0; kk < KDIM; kk += 32) {
            v16h afrag[2], bfrag[2];
            const int ka = kk + hi16 * 8;   // A: halves at K {ka..ka+7, ka+16..ka+23}
            #pragma unroll
            for (int tm = 0; tm < 2; ++tm) {
                const _Float16* ap = As + (rm + tm * 16 + nrow) * AK + ka;
                v8h lo = *(const v8h*)ap;
                v8h hg = *(const v8h*)(ap + 16);
                afrag[tm] = __builtin_shufflevector(lo, hg,
                    0,1,2,3,4,5,6,7,8,9,10,11,12,13,14,15);
            }
            const int kb = kk + hi16 * 8;   // B: halves at K {kb..kb+7, kb+16..kb+23}
            #pragma unroll
            for (int tn = 0; tn < 2; ++tn) {
                const _Float16* bp = Bsh + (pn + tn * 16 + nrow) * AK + kb;
                v8h lo = *(const v8h*)bp;
                v8h hg = *(const v8h*)(bp + 16);
                bfrag[tn] = __builtin_shufflevector(lo, hg,
                    0,1,2,3,4,5,6,7,8,9,10,11,12,13,14,15);
            }
            #pragma unroll
            for (int tm = 0; tm < 2; ++tm)
                #pragma unroll
                for (int tn = 0; tn < 2; ++tn)
                    acc[tm][tn] = __builtin_amdgcn_wmma_f32_16x16x32_f16(
                        false, afrag[tm], false, bfrag[tn],
                        (short)0, acc[tm][tn], false, false);
        }

        // ---- Store: C layout -> VGPR r holds M = r + 8*hi16, N = nrow
        #pragma unroll
        for (int tm = 0; tm < 2; ++tm) {
            const int obase = o0 + rm + tm * 16 + hi16 * 8;
            #pragma unroll
            for (int tn = 0; tn < 2; ++tn) {
                const int p = p0 + pn + tn * 16 + nrow;
                #pragma unroll
                for (int r = 0; r < 8; ++r) {
                    out[(size_t)(b * OC + obase + r) * POUT + p] =
                        acc[tm][tn][r] * scale;
                }
            }
        }
    }
}

extern "C" void kernel_launch(void* const* d_in, const int* in_sizes, int n_in,
                              void* d_out, int out_size, void* d_ws, size_t ws_size,
                              hipStream_t stream) {
    const float* x = (const float*)d_in[0];   // (32,128,4096) f32
    const float* w = (const float*)d_in[1];   // (256,128,1,2) f32
    float* out = (float*)d_out;               // (32,256,1024) f32

    dim3 grid(BS * (POUT / NT));              // 32 * 8 = 256 workgroups
    dim3 block(256);                          // 8 wave32 waves
    size_t smem = (size_t)(MT + NT) * AK * sizeof(_Float16);  // ~101 KB dynamic LDS
    conv_pairsum_wmma_kernel<<<grid, block, smem, stream>>>(x, w, out);
}
